// GAT_23948737643277
// MI455X (gfx1250) — compile-verified, matrix operand order checked by reading, not run on hardware
//
#include <hip/hip_runtime.h>
#include <hip/hip_fp16.h>

typedef __attribute__((ext_vector_type(16))) _Float16 v16h;
typedef __attribute__((ext_vector_type(8)))  _Float16 v8h;
typedef __attribute__((ext_vector_type(8)))  float    v8f;

#define N_NODES   50000
#define FIN       128      // in-features for every layer's GEMM (IN = HEADS*HID = 128)
#define GHEADS    2
#define NEG_SLOPE 0.2f

// ---------------------------------------------------------------- utilities

__global__ void gat_zero_f32(float* __restrict__ p, int n) {
  int i = blockIdx.x * blockDim.x + threadIdx.x;
  if (i < n) p[i] = 0.0f;
}

__global__ void gat_f32_to_f16(const float* __restrict__ src, _Float16* __restrict__ dst, int n) {
  int i = blockIdx.x * blockDim.x + threadIdx.x;
  if (i < n) dst[i] = (_Float16)src[i];
}

// ordered-uint encoding of float so unsigned atomicMax == float max
__device__ __forceinline__ unsigned gat_f2ord(float f) {
  unsigned u = __float_as_uint(f);
  return (u & 0x80000000u) ? ~u : (u | 0x80000000u);
}
__device__ __forceinline__ float gat_ord2f(unsigned u) {
  return __uint_as_float((u & 0x80000000u) ? (u & 0x7FFFFFFFu) : ~u);
}

// ---------------------------------------------------------------- W pre-pack
// Convert W (f32, row-major [128, Fout]) into B-fragment order:
// Bp[((tn*4 + kc)*32 + lane)*16 + i] = (f16) W[(kc*32 + K(lane,i)) * Fout + tn*16 + (lane&15)]
// so each GEMM lane reads its whole fragment as one contiguous 32B v16h.
__global__ void gat_pack_w(const float* __restrict__ W, _Float16* __restrict__ Bp, int Fout) {
  int t = blockIdx.x * blockDim.x + threadIdx.x;
  if (t >= Fout * FIN) return;
  int i    = t & 15;
  int lane = (t >> 4) & 31;
  int kc   = (t >> 9) & 3;
  int tn   = t >> 11;
  int half = lane >> 4, l15 = lane & 15, Kb = half * 8;
  int K    = (i < 8) ? (Kb + i) : (16 + Kb + (i - 8));
  Bp[t] = (_Float16)W[(size_t)(kc * 32 + K) * Fout + tn * 16 + l15];
}

// ---------------------------------------------------------------- WMMA GEMM
// feat[M, Fout] = A[M, 128] (f16) * W (pre-packed f16 fragments), f32 accumulate.
// One wave owns a 16-row strip: loads 4 A fragments once, then sweeps all
// Fout/16 output tiles with 4x v_wmma_f32_16x16x32_f16 each.
template <int FOUT>
__global__ __launch_bounds__(128)
void gat_gemm_wmma(const _Float16* __restrict__ A, const _Float16* __restrict__ Bp,
                   float* __restrict__ C, int M) {
  const int lane  = threadIdx.x & 31;
  const int wave  = threadIdx.x >> 5;
  const int strip = blockIdx.x * 4 + wave;
  if (strip * 16 >= M) return;

  const int M0   = strip * 16;
  const int half = lane >> 4;   // 0: lanes 0-15, 1: lanes 16-31
  const int l15  = lane & 15;
  const int Kb   = half * 8;

  // A fragments: lane holds row M=l15, K in [Kb..Kb+7] and [16+Kb..16+Kb+7]
  const _Float16* arow = A + (size_t)(M0 + l15) * FIN;
  v16h a[4];
  #pragma unroll
  for (int kc = 0; kc < 4; ++kc) {
    v8h lo = *(const v8h*)(arow + kc * 32 + Kb);
    v8h hi = *(const v8h*)(arow + kc * 32 + Kb + 16);
    #pragma unroll
    for (int i = 0; i < 8; ++i) { a[kc][i] = lo[i]; a[kc][i + 8] = hi[i]; }
  }

  #pragma unroll
  for (int tn = 0; tn < FOUT / 16; ++tn) {
    v8f c = {};
    #pragma unroll
    for (int kc = 0; kc < 4; ++kc) {
      v16h b = *(const v16h*)(Bp + ((size_t)(tn * 4 + kc) * 32 + lane) * 16);
      c = __builtin_amdgcn_wmma_f32_16x16x32_f16(
          /*neg_a=*/false, a[kc], /*neg_b=*/false, b,
          /*c_mod=*/(short)0, c, /*reuse_a=*/false, /*reuse_b=*/false);
    }
    // C/D layout: VGPR r -> row M0 + r + 8*half, col tn*16 + l15
    #pragma unroll
    for (int r = 0; r < 8; ++r)
      C[(size_t)(M0 + r + 8 * half) * FOUT + tn * 16 + l15] = c[r];
  }
}

// ---------------------------------------------------------------- per-node alphas
__global__ void gat_alpha(const float* __restrict__ feat,
                          const float* __restrict__ a_src, const float* __restrict__ a_dst,
                          float* __restrict__ alphas, float* __restrict__ alphad, int C) {
  int n = blockIdx.x * blockDim.x + threadIdx.x;
  if (n >= N_NODES) return;
  const float* f = feat + (size_t)n * (GHEADS * C);
  #pragma unroll
  for (int h = 0; h < GHEADS; ++h) {
    float ss = 0.0f, sd = 0.0f;
    for (int cc = 0; cc < C; ++cc) {
      float v = f[h * C + cc];
      ss += v * a_src[h * C + cc];
      sd += v * a_dst[h * C + cc];
    }
    alphas[n * GHEADS + h] = ss;
    alphad[n * GHEADS + h] = sd;
  }
}

// ---------------------------------------------------------------- edge passes
// Pass 1: e = leakyrelu(as[src]+ad[dst]); store to ew; segment max via ordered atomicMax.
__global__ void gat_edge_max(const int* __restrict__ ei, int E,
                             const float* __restrict__ alphas, const float* __restrict__ alphad,
                             float* __restrict__ ew, unsigned* __restrict__ mmax) {
  int e = blockIdx.x * blockDim.x + threadIdx.x;
  int ET = E + N_NODES;
  if (e >= ET) return;
  int s, d;
  if (e < E) { s = ei[e]; d = ei[E + e]; } else { s = d = e - E; }
  #pragma unroll
  for (int h = 0; h < GHEADS; ++h) {
    float v = alphas[s * GHEADS + h] + alphad[d * GHEADS + h];
    v = (v >= 0.0f) ? v : NEG_SLOPE * v;
    ew[(size_t)e * GHEADS + h] = v;
    atomicMax(&mmax[d * GHEADS + h], gat_f2ord(v));
  }
}

// Pass 2: w = exp(e - m[dst]); overwrite ew; segment sum into denom.
__global__ void gat_edge_exp(const int* __restrict__ ei, int E,
                             const unsigned* __restrict__ mmax,
                             float* __restrict__ ew, float* __restrict__ denom) {
  int e = blockIdx.x * blockDim.x + threadIdx.x;
  int ET = E + N_NODES;
  if (e >= ET) return;
  int d;
  if (e < E) { d = ei[E + e]; } else { d = e - E; }
  #pragma unroll
  for (int h = 0; h < GHEADS; ++h) {
    float w = expf(ew[(size_t)e * GHEADS + h] - gat_ord2f(mmax[d * GHEADS + h]));
    ew[(size_t)e * GHEADS + h] = w;
    atomicAdd(&denom[d * GHEADS + h], w);
  }
}

// Pass 3: acc[dst] += (w/denom[dst]) * feat[src], one thread per (edge, float4 chunk).
__global__ void gat_edge_aggregate(const int* __restrict__ ei, int E,
                                   const float* __restrict__ ew, const float* __restrict__ denom,
                                   const float* __restrict__ feat, float* __restrict__ acc, int C) {
  const int F = GHEADS * C;
  const int chunks = F / 4;
  long long t = (long long)blockIdx.x * blockDim.x + threadIdx.x;
  long long ET = (long long)E + N_NODES;
  if (t >= ET * chunks) return;
  int e = (int)(t / chunks);
  int j = (int)(t % chunks);
  int s, d;
  if (e < E) { s = ei[e]; d = ei[E + e]; } else { s = d = e - E; }
  int h = (j * 4) / C;
  float alpha = ew[(size_t)e * GHEADS + h] / (denom[d * GHEADS + h] + 1e-16f);
  const float4 hv = *(const float4*)(feat + (size_t)s * F + j * 4);
  float* o = acc + (size_t)d * F + j * 4;
  atomicAdd(o + 0, alpha * hv.x);
  atomicAdd(o + 1, alpha * hv.y);
  atomicAdd(o + 2, alpha * hv.z);
  atomicAdd(o + 3, alpha * hv.w);
}

// ---------------------------------------------------------------- finalize
__global__ void gat_finalize_elu(float* __restrict__ acc, const float* __restrict__ bias, int F) {
  int i = blockIdx.x * blockDim.x + threadIdx.x;
  if (i >= N_NODES * F) return;
  float v = acc[i] + bias[i % F];
  acc[i] = (v > 0.0f) ? v : (expf(v) - 1.0f);
}

__global__ void gat_finalize_mean(const float* __restrict__ acc, const float* __restrict__ bias,
                                  float* __restrict__ out, int C) {
  int i = blockIdx.x * blockDim.x + threadIdx.x;   // over N*C
  if (i >= N_NODES * C) return;
  int n = i / C, o = i % C;
  float v = 0.0f;
  #pragma unroll
  for (int h = 0; h < GHEADS; ++h) v += acc[(size_t)n * GHEADS * C + h * C + o];
  out[i] = v * (1.0f / GHEADS) + bias[o];
}

// ---------------------------------------------------------------- launch
extern "C" void kernel_launch(void* const* d_in, const int* in_sizes, int n_in,
                              void* d_out, int out_size, void* d_ws, size_t ws_size,
                              hipStream_t stream) {
  (void)n_in; (void)out_size; (void)ws_size;

  const float* x  = (const float*)d_in[0];
  const int*   ei = (const int*)d_in[1];
  const float* W[3]  = { (const float*)d_in[2],  (const float*)d_in[6],  (const float*)d_in[10] };
  const float* AS[3] = { (const float*)d_in[3],  (const float*)d_in[7],  (const float*)d_in[11] };
  const float* AD[3] = { (const float*)d_in[4],  (const float*)d_in[8],  (const float*)d_in[12] };
  const float* BI[3] = { (const float*)d_in[5],  (const float*)d_in[9],  (const float*)d_in[13] };

  const int E  = in_sizes[1] / 2;
  const int ET = E + N_NODES;

  // workspace carve-out (256B aligned)
  char*  ws  = (char*)d_ws;
  size_t off = 0;
  auto alloc = [&](size_t bytes) -> void* {
    off = (off + 255) & ~(size_t)255;
    void* p = ws + off;
    off += bytes;
    return p;
  };
  _Float16* xh     = (_Float16*)alloc((size_t)N_NODES * FIN * sizeof(_Float16));
  _Float16* wpack  = (_Float16*)alloc((size_t)FIN * FIN * sizeof(_Float16));
  float*    feat   = (float*)   alloc((size_t)N_NODES * FIN * sizeof(float));
  float*    acc    = (float*)   alloc((size_t)N_NODES * FIN * sizeof(float));
  float*    alphas = (float*)   alloc((size_t)N_NODES * GHEADS * sizeof(float));
  float*    alphad = (float*)   alloc((size_t)N_NODES * GHEADS * sizeof(float));
  unsigned* mmax   = (unsigned*)alloc((size_t)N_NODES * GHEADS * sizeof(unsigned));
  float*    denom  = (float*)   alloc((size_t)N_NODES * GHEADS * sizeof(float));
  float*    ew     = (float*)   alloc((size_t)ET * GHEADS * sizeof(float));

  auto cdiv = [](long long a, long long b) -> int { return (int)((a + b - 1) / b); };

  auto run_layer = [&](const float* xin, int layer, int Fout, int C, bool last) {
    // 1) convert input to f16; pack weights (f32 -> f16, fragment order)
    gat_f32_to_f16<<<cdiv((long long)N_NODES * FIN, 256), 256, 0, stream>>>(xin, xh, N_NODES * FIN);
    gat_pack_w<<<cdiv((long long)FIN * Fout, 256), 256, 0, stream>>>(W[layer], wpack, Fout);
    // 2) GEMM: feat = xh @ W
    dim3 ggrid(cdiv(N_NODES / 16, 4));
    if (Fout == 128) gat_gemm_wmma<128><<<ggrid, 128, 0, stream>>>(xh, wpack, feat, N_NODES);
    else             gat_gemm_wmma< 32><<<ggrid, 128, 0, stream>>>(xh, wpack, feat, N_NODES);
    // 3) per-node attention logits
    gat_alpha<<<cdiv(N_NODES, 256), 256, 0, stream>>>(feat, AS[layer], AD[layer], alphas, alphad, C);
    // 4) clear accumulators (acc was consumed as xin already for layers 1,2)
    gat_zero_f32<<<cdiv((long long)N_NODES * Fout, 256), 256, 0, stream>>>(acc, N_NODES * Fout);
    gat_zero_f32<<<cdiv((long long)N_NODES * GHEADS, 256), 256, 0, stream>>>(denom, N_NODES * GHEADS);
    gat_zero_f32<<<cdiv((long long)N_NODES * GHEADS, 256), 256, 0, stream>>>((float*)mmax, N_NODES * GHEADS);
    // 5) edge softmax + aggregate
    gat_edge_max<<<cdiv(ET, 256), 256, 0, stream>>>(ei, E, alphas, alphad, ew, mmax);
    gat_edge_exp<<<cdiv(ET, 256), 256, 0, stream>>>(ei, E, mmax, ew, denom);
    long long aggT = (long long)ET * (GHEADS * C / 4);
    gat_edge_aggregate<<<cdiv(aggT, 256), 256, 0, stream>>>(ei, E, ew, denom, feat, acc, C);
    // 6) finalize
    if (!last) {
      gat_finalize_elu<<<cdiv((long long)N_NODES * Fout, 256), 256, 0, stream>>>(acc, BI[layer], Fout);
    } else {
      gat_finalize_mean<<<cdiv((long long)N_NODES * C, 256), 256, 0, stream>>>(acc, BI[layer], (float*)d_out, C);
    }
  };

  run_layer(x,   0, 128, 64, false);  // layer 0: 128 -> 2x64 concat, ELU
  run_layer(acc, 1, 128, 64, false);  // layer 1: 128 -> 2x64 concat, ELU
  run_layer(acc, 2,  32, 16, true);   // layer 2: 128 -> 2x16, head-mean -> d_out
}
